// EBP_binaryNet_29283087024617
// MI455X (gfx1250) — matrix operand order repaired and checked by reading, not compiled, not run on hardware
//
#include <hip/hip_runtime.h>
#include <hip/hip_bf16.h>

// ---------------------------------------------------------------------------
// EBP binaryNet forward on gfx1250 (MI455X), WMMA f16 path.
// Wave tiles are 32(M) x 64(N): 2 A-fragments share 4 B-fragments,
// 24 bytes loaded per v_wmma_f32_16x16x32_f16 (16384 flops) in the fused layer.
// ---------------------------------------------------------------------------

typedef __attribute__((ext_vector_type(16))) _Float16 v16h;
typedef __attribute__((ext_vector_type(8)))  _Float16 v8h;
typedef __attribute__((ext_vector_type(8)))  float    v8f;

#define SQ2PI 0.79788456f
#define HDIM  2048      // hidden width
#define MBAT  2048      // batch
#define DIN   784       // input features
#define KP    800       // DIN padded to multiple of 32
#define DOUT  10        // classes
#define NPAD  16        // DOUT padded to 16

// ---------------------------------------------------------------------------
// Fragment loaders (layouts per CDNA5 ISA 7.12.2)
// A 16x32 f16: lane l holds row (l&15); half h -> k = (h&7) + 16*(h>>3) + 8*(l>=16)
//   => two contiguous 8-half (16B) loads per lane.
// B 32x16 f16 (from transposed weight BT[N][K]): lane l holds col (l&15);
//   half h -> k = (h&7) + 16*(h>>3) + 8*(l>=16)  => two contiguous 8-half (16B) loads.
// ---------------------------------------------------------------------------
__device__ __forceinline__ v16h load_frag_a(const _Float16* __restrict__ p,
                                            int rowBase, int ld, int kb, int lane) {
    const _Float16* base = p + (size_t)(rowBase + (lane & 15)) * ld + kb + ((lane >> 4) << 3);
    union { v16h v; v8h h[2]; } u;
    u.h[0] = *(const v8h*)(base);
    u.h[1] = *(const v8h*)(base + 16);
    return u.v;
}

__device__ __forceinline__ v16h load_frag_b(const _Float16* __restrict__ pT,
                                            int colBase, int ld, int kb, int lane) {
    const _Float16* base = pT + (size_t)(colBase + (lane & 15)) * ld + kb + ((lane >> 4) << 3);
    union { v16h v; v8h h[2]; } u;
    u.h[0] = *(const v8h*)(base);
    u.h[1] = *(const v8h*)(base + 16);
    return u.v;
}

__device__ __forceinline__ v8f wmma_f16(v16h a, v16h b, v8f c) {
    return __builtin_amdgcn_wmma_f32_16x16x32_f16(false, a, false, b, (short)0, c, false, false);
}

// ---------------------------------------------------------------------------
// Prep kernels
// ---------------------------------------------------------------------------
__global__ void k_pad_x(const float* __restrict__ x, _Float16* __restrict__ xp) {
    int idx = blockIdx.x * blockDim.x + threadIdx.x;
    if (idx >= MBAT * KP) return;
    int b = idx / KP, i = idx % KP;
    xp[idx] = (i < DIN) ? (_Float16)x[(size_t)b * DIN + i] : (_Float16)0.0f;
}

// m0T[j][i] = tanh(0.5*w0[i][j]), j<HDIM, i<KP (zero padded past DIN)
__global__ void k_prep_m0T(const float* __restrict__ w0, _Float16* __restrict__ m0T) {
    int idx = blockIdx.x * blockDim.x + threadIdx.x;
    if (idx >= HDIM * KP) return;
    int j = idx / KP, i = idx % KP;
    float t = (i < DIN) ? tanhf(0.5f * w0[(size_t)i * HDIM + j]) : 0.0f;
    m0T[idx] = (_Float16)t;
}

// mT[j][i] = tanh(0.5*w[i][j]);  msqT = mT^2   (HDIM x HDIM)
__global__ void k_prep_mT_sq(const float* __restrict__ w,
                             _Float16* __restrict__ mT, _Float16* __restrict__ msqT) {
    int idx = blockIdx.x * blockDim.x + threadIdx.x;
    if (idx >= HDIM * HDIM) return;
    int j = idx >> 11, i = idx & (HDIM - 1);
    float t = tanhf(0.5f * w[(size_t)i * HDIM + j]);
    mT[idx]   = (_Float16)t;
    msqT[idx] = (_Float16)(t * t);
}

// mlastT[j][i], j in [0,16): rows >= DOUT zeroed
__global__ void k_prep_mlastT(const float* __restrict__ wl, _Float16* __restrict__ mlT) {
    int idx = blockIdx.x * blockDim.x + threadIdx.x;
    if (idx >= NPAD * HDIM) return;
    int j = idx >> 11, i = idx & (HDIM - 1);
    float t = (j < DOUT) ? tanhf(0.5f * wl[(size_t)i * DOUT + j]) : 0.0f;
    mlT[idx] = (_Float16)t;
}

// csum[j] = sum_{i<K} (1 - tanh(0.5*w[i][j])^2)
__global__ void k_csum(const float* __restrict__ w, float* __restrict__ cs, int K) {
    int j = blockIdx.x * blockDim.x + threadIdx.x;
    if (j >= HDIM) return;
    float s = 0.0f;
    for (int i = 0; i < K; ++i) {
        float t = tanhf(0.5f * w[(size_t)i * HDIM + j]);
        s += 1.0f - t * t;
    }
    cs[j] = s;
}

// ---------------------------------------------------------------------------
// Layer 1: h = SQ2PI*(x@m0 + th0)/sqrt(csum0); xbar = tanh(h); xcov = 1-xbar^2
// Wave computes a 32x64 tile (2 A-frags x 4 B-frags). grid(32,8) x block(256)
// ---------------------------------------------------------------------------
__global__ void __launch_bounds__(256)
k_layer1(const _Float16* __restrict__ xp, const _Float16* __restrict__ m0T,
         const float* __restrict__ csum, const float* __restrict__ th,
         _Float16* __restrict__ xb_out, _Float16* __restrict__ xcov_out) {
    int lane = threadIdx.x & 31, wave = threadIdx.x >> 5;
    int mBase = (blockIdx.y * 8 + wave) * 32;
    int nBase = blockIdx.x * 64;

    v8f acc[2][4] = {};
    for (int kb = 0; kb < KP; kb += 32) {
        v16h a0 = load_frag_a(xp, mBase,      KP, kb, lane);
        v16h a1 = load_frag_a(xp, mBase + 16, KP, kb, lane);
#pragma unroll
        for (int t = 0; t < 4; ++t) {
            v16h b = load_frag_b(m0T, nBase + t * 16, KP, kb, lane);
            acc[0][t] = wmma_f16(a0, b, acc[0][t]);
            acc[1][t] = wmma_f16(a1, b, acc[1][t]);
        }
    }
    int rowOff = (lane >> 4) * 8;
#pragma unroll
    for (int rt = 0; rt < 2; ++rt) {
#pragma unroll
        for (int t = 0; t < 4; ++t) {
            int n = nBase + t * 16 + (lane & 15);
            float rs = rsqrtf(csum[n]);
            float tb = th[n];
#pragma unroll
            for (int r = 0; r < 8; ++r) {
                int grow = mBase + rt * 16 + rowOff + r;
                float hv = SQ2PI * (acc[rt][t][r] + tb) * rs;
                float xb = tanhf(hv);
                size_t o = (size_t)grow * HDIM + n;
                xb_out[o]   = (_Float16)xb;
                xcov_out[o] = (_Float16)(1.0f - xb * xb);
            }
        }
    }
}

// ---------------------------------------------------------------------------
// Generic EBP layer: dot1 = xbar@m, dot2 = xcov@m^2
// dsig = dot2 + csum;  xb = tanh(SQ2PI*(dot1+th)/sqrt(dsig))
// Wave computes 32x64 of BOTH GEMMs (16 wmma / k-step).
// Writes xb (f16) and either f16 cov (mid layers) or f32 cov (final, to d_out)
// ---------------------------------------------------------------------------
__global__ void __launch_bounds__(256)
k_layer(const _Float16* __restrict__ xb_in, const _Float16* __restrict__ xc_in,
        const _Float16* __restrict__ mT, const _Float16* __restrict__ msqT,
        const float* __restrict__ csum, const float* __restrict__ th,
        _Float16* __restrict__ xb_out, _Float16* __restrict__ xcov16_out,
        float* __restrict__ xcov32_out) {
    int lane = threadIdx.x & 31, wave = threadIdx.x >> 5;
    int mBase = (blockIdx.y * 8 + wave) * 32;
    int nBase = blockIdx.x * 64;

    v8f acc1[2][4] = {};
    v8f acc2[2][4] = {};
    for (int kb = 0; kb < HDIM; kb += 32) {
        v16h a10 = load_frag_a(xb_in, mBase,      HDIM, kb, lane);
        v16h a11 = load_frag_a(xb_in, mBase + 16, HDIM, kb, lane);
        v16h a20 = load_frag_a(xc_in, mBase,      HDIM, kb, lane);
        v16h a21 = load_frag_a(xc_in, mBase + 16, HDIM, kb, lane);
#pragma unroll
        for (int t = 0; t < 4; ++t) {
            v16h b1 = load_frag_b(mT,   nBase + t * 16, HDIM, kb, lane);
            v16h b2 = load_frag_b(msqT, nBase + t * 16, HDIM, kb, lane);
            acc1[0][t] = wmma_f16(a10, b1, acc1[0][t]);
            acc1[1][t] = wmma_f16(a11, b1, acc1[1][t]);
            acc2[0][t] = wmma_f16(a20, b2, acc2[0][t]);
            acc2[1][t] = wmma_f16(a21, b2, acc2[1][t]);
        }
    }
    int rowOff = (lane >> 4) * 8;
#pragma unroll
    for (int rt = 0; rt < 2; ++rt) {
#pragma unroll
        for (int t = 0; t < 4; ++t) {
            int n = nBase + t * 16 + (lane & 15);
            float cs = csum[n];
            float tb = th[n];
#pragma unroll
            for (int r = 0; r < 8; ++r) {
                int grow = mBase + rt * 16 + rowOff + r;
                float dsig = acc2[rt][t][r] + cs;
                float hv = SQ2PI * (acc1[rt][t][r] + tb) * rsqrtf(dsig);
                float xb = tanhf(hv);
                size_t o = (size_t)grow * HDIM + n;
                xb_out[o] = (_Float16)xb;
                float cv = 1.0f - xb * xb;
                if (xcov16_out) xcov16_out[o] = (_Float16)cv;
                else            xcov32_out[o] = cv;
            }
        }
    }
}

// ---------------------------------------------------------------------------
// Last layer: hlast[2048][16] = x4bar @ mlast + thlast (cols >= DOUT junk-ok)
// Wave computes 32x16.
// ---------------------------------------------------------------------------
__global__ void __launch_bounds__(256)
k_last(const _Float16* __restrict__ x4, const _Float16* __restrict__ mlT,
       const float* __restrict__ thl, float* __restrict__ hlast) {
    int lane = threadIdx.x & 31, wave = threadIdx.x >> 5;
    int mBase = (blockIdx.x * 8 + wave) * 32;
    v8f acc[2] = {};
    for (int kb = 0; kb < HDIM; kb += 32) {
        v16h b  = load_frag_b(mlT, 0, HDIM, kb, lane);
        v16h a0 = load_frag_a(x4, mBase,      HDIM, kb, lane);
        v16h a1 = load_frag_a(x4, mBase + 16, HDIM, kb, lane);
        acc[0] = wmma_f16(a0, b, acc[0]);
        acc[1] = wmma_f16(a1, b, acc[1]);
    }
    int n = lane & 15;
    float tb = (n < DOUT) ? thl[n] : 0.0f;
    int rowOff = (lane >> 4) * 8;
#pragma unroll
    for (int rt = 0; rt < 2; ++rt) {
#pragma unroll
        for (int r = 0; r < 8; ++r) {
            hlast[(size_t)(mBase + rt * 16 + rowOff + r) * NPAD + n] = acc[rt][r] + tb;
        }
    }
}

// ---------------------------------------------------------------------------
// Per-row log-softmax, loss term, correctness flag
// ---------------------------------------------------------------------------
__global__ void k_softmax(const float* __restrict__ hlast, const int* __restrict__ target,
                          float* __restrict__ out_h, float* __restrict__ out_lp,
                          float* __restrict__ rloss, float* __restrict__ rcorr) {
    int b = blockIdx.x * blockDim.x + threadIdx.x;
    if (b >= MBAT) return;
    float v[DOUT];
    float mx = -3.402823e38f;
    int am = 0;
#pragma unroll
    for (int j = 0; j < DOUT; ++j) {
        v[j] = hlast[(size_t)b * NPAD + j];
        if (v[j] > mx) { mx = v[j]; am = j; }
    }
    float s = 0.0f;
#pragma unroll
    for (int j = 0; j < DOUT; ++j) s += expf(v[j] - mx);
    float lse = mx + logf(s);
#pragma unroll
    for (int j = 0; j < DOUT; ++j) {
        out_h[(size_t)b * DOUT + j]  = v[j];
        out_lp[(size_t)b * DOUT + j] = v[j] - lse;
    }
    int t = target[b];
    rloss[b] = lse - v[t];
    rcorr[b] = (am == t) ? 1.0f : 0.0f;
}

// Single-block deterministic reduction for the two scalars
__global__ void k_reduce(const float* __restrict__ rloss, const float* __restrict__ rcorr,
                         float* __restrict__ o_loss, float* __restrict__ o_fc) {
    __shared__ float s1[256];
    __shared__ float s2[256];
    float a = 0.0f, c = 0.0f;
    for (int i = threadIdx.x; i < MBAT; i += 256) { a += rloss[i]; c += rcorr[i]; }
    s1[threadIdx.x] = a; s2[threadIdx.x] = c;
    __syncthreads();
    for (int st = 128; st > 0; st >>= 1) {
        if (threadIdx.x < st) {
            s1[threadIdx.x] += s1[threadIdx.x + st];
            s2[threadIdx.x] += s2[threadIdx.x + st];
        }
        __syncthreads();
    }
    if (threadIdx.x == 0) {
        o_loss[0] = s1[0] / (float)MBAT;
        o_fc[0]   = s2[0] / (float)MBAT;
    }
}

// ---------------------------------------------------------------------------
extern "C" void kernel_launch(void* const* d_in, const int* in_sizes, int n_in,
                              void* d_out, int out_size, void* d_ws, size_t ws_size,
                              hipStream_t stream) {
    const float* x      = (const float*)d_in[0];
    const int*   target = (const int*)d_in[1];
    const float* w0     = (const float*)d_in[2];
    const float* w1     = (const float*)d_in[3];
    // d_in[4] = w2 (unused by the reference forward)
    const float* w3     = (const float*)d_in[5];
    const float* wlast  = (const float*)d_in[6];
    const float* th0    = (const float*)d_in[7];
    const float* th1    = (const float*)d_in[8];
    const float* th2    = (const float*)d_in[9];
    const float* thlast = (const float*)d_in[10];

    char* ws = (char*)d_ws;
    size_t off = 0;
    auto alloc = [&](size_t bytes) -> char* {
        char* p = ws + off;
        off = (off + bytes + 255) & ~(size_t)255;
        return p;
    };
    _Float16* xp     = (_Float16*)alloc((size_t)MBAT * KP * 2);
    _Float16* m0T    = (_Float16*)alloc((size_t)HDIM * KP * 2);
    _Float16* m1T    = (_Float16*)alloc((size_t)HDIM * HDIM * 2);
    _Float16* m1sqT  = (_Float16*)alloc((size_t)HDIM * HDIM * 2);
    _Float16* m3T    = (_Float16*)alloc((size_t)HDIM * HDIM * 2);
    _Float16* m3sqT  = (_Float16*)alloc((size_t)HDIM * HDIM * 2);
    _Float16* mlT    = (_Float16*)alloc((size_t)NPAD * HDIM * 2);
    float*    csum0  = (float*)alloc(HDIM * 4);
    float*    csum1  = (float*)alloc(HDIM * 4);
    float*    csum3  = (float*)alloc(HDIM * 4);
    _Float16* x1b    = (_Float16*)alloc((size_t)MBAT * HDIM * 2);
    _Float16* xc1    = (_Float16*)alloc((size_t)MBAT * HDIM * 2);
    _Float16* x2b    = (_Float16*)alloc((size_t)MBAT * HDIM * 2);
    _Float16* xc2    = (_Float16*)alloc((size_t)MBAT * HDIM * 2);
    _Float16* x4b    = (_Float16*)alloc((size_t)MBAT * HDIM * 2);
    float*    hlast  = (float*)alloc((size_t)MBAT * NPAD * 4);
    float*    rloss  = (float*)alloc(MBAT * 4);
    float*    rcorr  = (float*)alloc(MBAT * 4);

    float* out    = (float*)d_out;
    float* o_h    = out;                                  // [2048,10]
    float* o_lp   = out + (size_t)MBAT * DOUT;            // [2048,10]
    float* o_cov  = out + (size_t)2 * MBAT * DOUT;        // [2048,2048]
    float* o_loss = o_cov + (size_t)MBAT * HDIM;          // scalar
    float* o_fc   = o_loss + 1;                           // scalar

    // ---- prep ----
    k_pad_x<<<(MBAT * KP + 255) / 256, 256, 0, stream>>>(x, xp);
    k_prep_m0T<<<(HDIM * KP + 255) / 256, 256, 0, stream>>>(w0, m0T);
    k_csum<<<HDIM / 256, 256, 0, stream>>>(w0, csum0, DIN);
    k_prep_mT_sq<<<(HDIM * HDIM) / 256, 256, 0, stream>>>(w1, m1T, m1sqT);
    k_csum<<<HDIM / 256, 256, 0, stream>>>(w1, csum1, HDIM);
    k_prep_mT_sq<<<(HDIM * HDIM) / 256, 256, 0, stream>>>(w3, m3T, m3sqT);
    k_csum<<<HDIM / 256, 256, 0, stream>>>(w3, csum3, HDIM);
    k_prep_mlastT<<<(NPAD * HDIM) / 256, 256, 0, stream>>>(wlast, mlT);

    // ---- layers (WMMA GEMMs), 32x64 wave tiles ----
    dim3 g(HDIM / 64, MBAT / 256);
    k_layer1<<<g, 256, 0, stream>>>(xp, m0T, csum0, th0, x1b, xc1);
    k_layer<<<g, 256, 0, stream>>>(x1b, xc1, m1T, m1sqT, csum1, th1, x2b, xc2, (float*)nullptr);
    k_layer<<<g, 256, 0, stream>>>(x2b, xc2, m3T, m3sqT, csum3, th2, x4b, (_Float16*)nullptr, o_cov);
    k_last<<<MBAT / 256, 256, 0, stream>>>(x4b, mlT, thlast, hlast);

    // ---- outputs ----
    k_softmax<<<MBAT / 256, 256, 0, stream>>>(hlast, target, o_h, o_lp, rloss, rcorr);
    k_reduce<<<1, 256, 0, stream>>>(rloss, rcorr, o_loss, o_fc);
    (void)in_sizes; (void)n_in; (void)out_size; (void)ws_size;
}
